// Spectrogram_13924283974388
// MI455X (gfx1250) — matrix operand order repaired and checked by reading, not compiled, not run on hardware
//
#include <hip/hip_runtime.h>

typedef __attribute__((ext_vector_type(16))) _Float16 v16h;
typedef __attribute__((ext_vector_type(8)))  float    v8f;
typedef __attribute__((ext_vector_type(4)))  float    v4f;

#define FRAME_LEN      512
#define FRAME_STEP     256
#define NBINS          257
#define PAIRS_PER_WAVE 4
#define NTILES_HALF    20            // pad 257 bins -> 320 cols (20 tiles) per half
#define NTILES         40            // real half + imag half
#define SLOTS          5             // 20 tiles / 4 pairs per wave
#define KTILES         16            // 512 / 32
#define BATCH          64
#define TFRAMES        1874          // (480000-512)/256 + 1
#define MROWS          (BATCH * TFRAMES)   // 119936 = 7496 * 16 exactly
#define MTILES         (MROWS / 16)        // 7496
#define EPSF           1e-7f
#define FRAG_BYTES     1024          // one B fragment: 32 lanes * 32 B

// ---------------------------------------------------------------------------
// Prep: fold symmetric Hann window + density scaling into the DFT weights,
// convert to f16, pack into WMMA B-fragment layout, PAIR-MAJOR:
//   frag(nt, kt) at byte offset (nt*KTILES + kt)*1024, lane offset lane*32
//   element h = W[32*kt + (lane>=16 ? 8 : 0) + (h<8 ? h : h+8)][16*ntl + (lane&15)]
// (B 32x16 f16: lanes 0-15 hold K=0..7,16..23, lanes 16-31 hold K=8..15,24..31, n = lane&15)
// Padded columns (bin >= 257) are zero.
// ---------------------------------------------------------------------------
__global__ void prep_weights(const float* __restrict__ realk,
                             const float* __restrict__ imagk,
                             _Float16* __restrict__ wpk) {
    int tid = blockIdx.x * blockDim.x + threadIdx.x;
    if (tid >= NTILES * KTILES * 32) return;
    int lane = tid & 31;
    int idx  = tid >> 5;
    int kt   = idx % KTILES;
    int nt   = idx / KTILES;

    const float* src = (nt < NTILES_HALF) ? realk : imagk;
    int ntl   = (nt < NTILES_HALF) ? nt : nt - NTILES_HALF;
    int col   = ntl * 16 + (lane & 15);
    int kbase = kt * 32 + ((lane >= 16) ? 8 : 0);

    // sum(hann_sym(512)^2) = 0.25*(512 - 2*1 + 256.5) = 191.625 (analytic)
    const float inv_scale = rsqrtf(191.625f + EPSF);
    const float two_pi_over = 6.283185307179586f / (float)(FRAME_LEN - 1);

    v16h w;
    #pragma unroll
    for (int h = 0; h < 16; ++h) {
        int k = kbase + (h < 8 ? h : h + 8);
        float win = 0.5f - 0.5f * cosf(two_pi_over * (float)k);
        float v = 0.0f;
        if (col < NBINS) v = src[k * NBINS + col] * win * inv_scale;
        w[h] = (_Float16)v;
    }
    *(v16h*)(wpk + (size_t)(idx * 32 + lane) * 16) = w;
}

// ---------------------------------------------------------------------------
// STFT-as-GEMM: one wave per (M-tile, slot); each wave owns 4 N-pairs =
// 8 independent f32 accumulator chains. Per k-step: one shared A fragment
// (f32 load + cvt to f16) feeds 8 v_wmma_f32_16x16x32_f16. Epilogue fuses
// power -> sqrt -> log with non-temporal stores (keep L2 for reused x).
// ---------------------------------------------------------------------------
__global__ __launch_bounds__(256) void stft_wmma(
    const float* __restrict__ x,
    const _Float16* __restrict__ wpk,
    float* __restrict__ out) {

    int wid   = blockIdx.x * 8 + (threadIdx.x >> 5);
    int lane  = threadIdx.x & 31;
    int mtile = wid / SLOTS;
    int slot  = wid % SLOTS;
    if (mtile >= MTILES) return;

    // Per-lane A row (A 16-bit 16x32 layout: row = lane&15 in both halves)
    int m = mtile * 16 + (lane & 15);
    int b = m / TFRAMES;
    int t = m - b * TFRAMES;
    // lane<16 covers K {0..7, 16..23} of each 32-chunk; lane>=16 covers +8
    const float* plane = x + (size_t)b * 480000 + (size_t)t * FRAME_STEP
                           + ((lane >> 4) << 3);

    // One base pointer for all 8 fragments per k-step; everything else is
    // a compile-time-constant byte offset (< 24-bit signed immediate).
    const char* wb = (const char*)wpk
                   + (size_t)slot * PAIRS_PER_WAVE * KTILES * FRAG_BYTES
                   + (size_t)lane * 32;

    v8f acr[PAIRS_PER_WAVE] = {};
    v8f aci[PAIRS_PER_WAVE] = {};

    #pragma unroll 2
    for (int kt = 0; kt < KTILES; ++kt) {
        const float* p0 = plane + kt * 32;
        v4f x0 = *(const v4f*)(p0);
        v4f x1 = *(const v4f*)(p0 + 4);
        v4f x2 = *(const v4f*)(p0 + 16);
        v4f x3 = *(const v4f*)(p0 + 20);
        v16h a;
        #pragma unroll
        for (int i = 0; i < 4; ++i) {
            a[i]      = (_Float16)x0[i];
            a[4 + i]  = (_Float16)x1[i];
            a[8 + i]  = (_Float16)x2[i];
            a[12 + i] = (_Float16)x3[i];
        }
        #pragma unroll
        for (int pp = 0; pp < PAIRS_PER_WAVE; ++pp) {
            v16h br = *(const v16h*)(wb + (pp * KTILES + kt) * FRAG_BYTES);
            v16h bi = *(const v16h*)(wb + NTILES_HALF * KTILES * FRAG_BYTES
                                        + (pp * KTILES + kt) * FRAG_BYTES);
            acr[pp] = __builtin_amdgcn_wmma_f32_16x16x32_f16(
                false, a, false, br, (short)0, acr[pp], false, false);
            aci[pp] = __builtin_amdgcn_wmma_f32_16x16x32_f16(
                false, a, false, bi, (short)0, aci[pp], false, false);
        }
    }

    // D layout: lane 0-15 -> M = r, lane 16-31 -> M = r + 8; N = lane & 15
    int nb    = slot * PAIRS_PER_WAVE * 16 + (lane & 15);
    int mrow0 = mtile * 16 + ((lane >> 4) << 3);
    #pragma unroll
    for (int pp = 0; pp < PAIRS_PER_WAVE; ++pp) {
        int n = nb + pp * 16;
        if (n < NBINS) {
            #pragma unroll
            for (int r = 0; r < 8; ++r) {
                float dr = acr[pp][r];
                float di = aci[pp][r];
                float pw = fmaf(dr, dr, fmaf(di, di, EPSF));
                __builtin_nontemporal_store(
                    logf(EPSF + sqrtf(pw)),
                    out + (size_t)(mrow0 + r) * NBINS + n);
            }
        }
    }
}

extern "C" void kernel_launch(void* const* d_in, const int* in_sizes, int n_in,
                              void* d_out, int out_size, void* d_ws, size_t ws_size,
                              hipStream_t stream) {
    (void)in_sizes; (void)n_in; (void)out_size; (void)ws_size;
    const float* x  = (const float*)d_in[0];
    const float* rk = (const float*)d_in[1];
    const float* ik = (const float*)d_in[2];
    float* out = (float*)d_out;
    _Float16* wpk = (_Float16*)d_ws;   // 40*16 fragments * 1 KB = 640 KB

    int prep_threads = NTILES * KTILES * 32;                 // 20480
    prep_weights<<<(prep_threads + 255) / 256, 256, 0, stream>>>(rk, ik, wpk);

    int nwaves = MTILES * SLOTS;                             // 37480 (exact /8)
    stft_wmma<<<(nwaves + 7) / 8, 256, 0, stream>>>(x, wpk, out);
}
